// SE3PointNeighborConvolution_78082505441809
// MI455X (gfx1250) — matrix operand order repaired and checked by reading, not compiled, not run on hardware
//
#include <hip/hip_runtime.h>
#include <hip/hip_bf16.h>
#include <stdint.h>

typedef __attribute__((ext_vector_type(16))) __bf16 v16bf;
typedef __attribute__((ext_vector_type(8)))  float  v8f;

#define NPTS   16384
#define GAMMA_ 10.0f

union BF16x16 { uint4 q[2]; __bf16 h[16]; v16bf v; };
union BF16x8  { uint4 q;    __bf16 h[8]; };

// ---------------------------------------------------------------------------
// Prologue 1: transpose input [16][N] f32 -> inT [N][16] bf16 (32B per point)
// ---------------------------------------------------------------------------
__global__ void prep_input_kernel(const float* __restrict__ in,
                                  __bf16* __restrict__ inT) {
    int n = blockIdx.x * blockDim.x + threadIdx.x;
    BF16x16 t;
#pragma unroll
    for (int i = 0; i < 16; ++i) t.h[i] = (__bf16)in[i * NPTS + n];
    uint4* dst = (uint4*)(inT + n * 16);
    dst[0] = t.q[0];
    dst[1] = t.q[1];
}

// ---------------------------------------------------------------------------
// Prologue 2: pre-swizzle W into the stage-2 A-operand register layout.
// Stage-2 contraction index t = b*16 + i (chosen so the G store in the main
// kernel is a single contiguous b128 per lane).  Chunk c covers t=c*32..+31;
// within a chunk this code's contiguous k order maps lane-element j -> kk:
//   lo lanes: j0-15 -> kk0-15 ; hi lanes: +16.
// ---------------------------------------------------------------------------
__global__ void prep_w_kernel(const float* __restrict__ W,
                              __bf16* __restrict__ wfrag) {
    int t    = threadIdx.x;       // 256 threads: 8 chunks x 32 lanes
    int c    = t >> 5;
    int lane = t & 31;
    int hf   = lane >> 4;
    int o    = lane & 15;
#pragma unroll
    for (int j = 0; j < 16; ++j) {
        int kk = j + 16 * hf;
        int tt = c * 32 + kk;          // global contraction index
        int i  = tt & 15;
        int b  = tt >> 4;
        wfrag[(c * 32 + lane) * 16 + j] = (__bf16)W[o * 256 + i * 16 + b];
    }
}

// ---------------------------------------------------------------------------
// Main: one wave per 16-point tile.
// Stage 1 (per point): G[16i x 16b] = A[16i x 32k] * B[32k x 16b]  (1 WMMA)
//   A = gathered neighbor features, B = mask * exp(-gamma*(r-center)^2)
// Stage 2 (per tile):  out[16o x 16p] = W2[16 x 256] * G[256 x 16p] (8 WMMA)
// ---------------------------------------------------------------------------
__global__ void __launch_bounds__(32)
se3_conv_kernel(const float* __restrict__ coords,
                const float* __restrict__ centers,
                const float* __restrict__ mask,
                const int*   __restrict__ nbrs,
                const __bf16* __restrict__ inT,
                const __bf16* __restrict__ wfrag,
                float* __restrict__ out) {
    __shared__ __align__(16) unsigned char s_x[16 * 80];   // [i][k] bf16, 80B row (bank-padded)
    __shared__ __align__(16) float         s_rm[64];       // 32 x (r, mask)
    __shared__ __align__(16) unsigned char s_g[16 * 528];  // [p][t = b*16+i] bf16, 528B row

    const int lane = threadIdx.x;
    const int hf   = lane >> 4;   // wave half
    const int col  = lane & 15;   // matrix column / row index within half
    const int n0   = blockIdx.x * 16;

    const float cb = centers[col];   // one RBF center per B-column lane

    for (int p = 0; p < 16; ++p) {
        const int n = n0 + p;
        // ---- per-neighbor work: lane == neighbor slot k ----
        const int   nb = nbrs[n * 32 + lane];
        const float m  = mask[n * 32 + lane];
        const float dx = coords[nb * 3 + 0] - coords[n * 3 + 0];
        const float dy = coords[nb * 3 + 1] - coords[n * 3 + 1];
        const float dz = coords[nb * 3 + 2] - coords[n * 3 + 2];
        const float r  = sqrtf(dx * dx + dy * dy + dz * dz + 1e-12f);
        s_rm[lane * 2 + 0] = r;
        s_rm[lane * 2 + 1] = m;

        // gather neighbor feature row (32B) and scatter transposed into LDS
        BF16x16 xr;
        const uint4* xp = (const uint4*)(inT + nb * 16);
        xr.q[0] = xp[0];
        xr.q[1] = xp[1];
#pragma unroll
        for (int i = 0; i < 16; ++i)
            ((__bf16*)(s_x + i * 80))[lane] = xr.h[i];
        __syncthreads();

        // ---- A fragment: row i = col; K halves per A-operand layout ----
        BF16x16 a;
        const unsigned char* arow = s_x + col * 80 + hf * 32;
        a.q[0] = *(const uint4*)(arow);        // lo: K0-7   | hi: K16-23
        a.q[1] = *(const uint4*)(arow + 16);   // lo: K8-15  | hi: K24-31

        // ---- B fragment: column b = col; element j -> k = j + 16*hf ----
        BF16x16 b;
#pragma unroll
        for (int jp = 0; jp < 8; ++jp) {
            const int    k0  = 2 * jp + 16 * hf;
            const float4 rm2 = *(const float4*)(s_rm + k0 * 2);  // (r,m)(r,m), broadcast read
            const float  d0  = rm2.x - cb;
            const float  d1  = rm2.z - cb;
            b.h[2 * jp + 0] = (__bf16)(rm2.y * __expf(-GAMMA_ * d0 * d0));
            b.h[2 * jp + 1] = (__bf16)(rm2.w * __expf(-GAMMA_ * d1 * d1));
        }

        v8f g = {};
        g = __builtin_amdgcn_wmma_f32_16x16x32_bf16(false, a.v, false, b.v,
                                                    (short)0, g, false, false);

        // park G[p] in LDS: lane's 8 values are contiguous at t = col*16 + i
        BF16x8 gb;
#pragma unroll
        for (int v = 0; v < 8; ++v) gb.h[v] = (__bf16)g[v];
        *(uint4*)(s_g + p * 528 + col * 32 + hf * 16) = gb.q;
        __syncthreads();
    }

    // loop-invariant stage-2 A fragments (pre-swizzled W), coalesced b128 loads
    v16bf wf[8];
#pragma unroll
    for (int c = 0; c < 8; ++c) {
        BF16x16 t;
        const uint4* wp = (const uint4*)(wfrag + (c * 32 + lane) * 16);
        t.q[0] = wp[0];
        t.q[1] = wp[1];
        wf[c] = t.v;
    }

    // ---- stage 2: 8 chained WMMAs over t = 0..255 ----
    v8f acc = {};
#pragma unroll
    for (int c = 0; c < 8; ++c) {
        BF16x16 b2;  // B column = point p = col; K = c*32 + j + 16*hf
        const unsigned char* grow = s_g + col * 528 + c * 64 + 32 * hf;
        b2.q[0] = *(const uint4*)(grow);
        b2.q[1] = *(const uint4*)(grow + 16);
        acc = __builtin_amdgcn_wmma_f32_16x16x32_bf16(false, wf[c], false, b2.v,
                                                      (short)0, acc, false, false);
    }
#pragma unroll
    for (int v = 0; v < 8; ++v) {
        const int o = v + 8 * hf;
        out[o * NPTS + n0 + col] = acc[v];
    }
}

extern "C" void kernel_launch(void* const* d_in, const int* in_sizes, int n_in,
                              void* d_out, int out_size, void* d_ws, size_t ws_size,
                              hipStream_t stream) {
    (void)in_sizes; (void)n_in; (void)out_size; (void)ws_size;
    const float* input   = (const float*)d_in[0];
    const float* coords  = (const float*)d_in[1];
    const float* W       = (const float*)d_in[2];
    const float* centers = (const float*)d_in[3];
    const float* maskp   = (const float*)d_in[4];
    const int*   nbrs    = (const int*)d_in[5];
    float* out = (float*)d_out;

    // workspace: inT bf16 [N][16] (512 KB) + wfrag bf16 [8][32][16] (8 KB)
    __bf16* inT   = (__bf16*)d_ws;
    __bf16* wfrag = (__bf16*)((char*)d_ws + (size_t)NPTS * 16 * 2);

    prep_input_kernel<<<NPTS / 256, 256, 0, stream>>>(input, inT);
    prep_w_kernel<<<1, 256, 0, stream>>>(W, wfrag);
    se3_conv_kernel<<<NPTS / 16, 32, 0, stream>>>(coords, centers, maskp, nbrs,
                                                  inT, wfrag, out);
}
